// MultiHeadAttention_23854248362716
// MI455X (gfx1250) — matrix operand order repaired and checked by reading, not compiled, not run on hardware
//
#include <hip/hip_runtime.h>
#include <stdint.h>

typedef __attribute__((ext_vector_type(16))) __bf16 v16bf;
typedef __attribute__((ext_vector_type(8)))  float  v8f;
typedef __attribute__((ext_vector_type(4)))  unsigned int v4u;
typedef __attribute__((ext_vector_type(8)))  int v8i_;
typedef __attribute__((ext_vector_type(4)))  int v4i_;

namespace {
constexpr int Bz = 2;
constexpr int S  = 2048;
constexpr int D  = 2048;
constexpr int H  = 16;
constexpr int HD = 128;
constexpr int M  = Bz * S;     // 4096 rows in the projection GEMMs
constexpr int QT = 128;        // q rows per attention block
constexpr int KT = 64;         // keys per streamed tile
// 1/sqrt(128) * log2(e): folded into Q so softmax runs in exp2 domain
constexpr float QSCALE = 0.088388347648318447f * 1.4426950408889634f;
}

#if __has_builtin(__builtin_amdgcn_tensor_load_to_lds) && \
    __has_builtin(__builtin_amdgcn_s_wait_tensorcnt)
#define HAS_TDM 1
#else
#define HAS_TDM 0
#endif

__device__ __forceinline__ v8f wmma_bf16(v16bf a, v16bf b, v8f c) {
  return __builtin_amdgcn_wmma_f32_16x16x32_bf16(false, a, false, b,
                                                 (short)0, c, false, false);
}

// Pack two f32 into a bf16x2 dword (lowers to v_cvt_pk_bf16_f32).
__device__ __forceinline__ unsigned pk_bf16(float lo, float hi) {
  union { __bf16 h[2]; unsigned u; } p;
  p.h[0] = (__bf16)lo;
  p.h[1] = (__bf16)hi;
  return p.u;
}

// A fragment: 16x32 bf16 tile, row-major source, leading dim ld (mult of 8).
// Lane l (m=l&15, hi=l>>4): elems 0..7 = K hi*8..hi*8+7, elems 8..15 = +16.
__device__ __forceinline__ v16bf load_a_frag(const __bf16* p, int ld) {
  const int lane = threadIdx.x & 31;
  const int m = lane & 15;
  const int hi = lane >> 4;
  union { v16bf v; uint4 q[2]; } u;
  const __bf16* row = p + m * ld + hi * 8;
  u.q[0] = *(const uint4*)(row);
  u.q[1] = *(const uint4*)(row + 16);
  return u.v;
}

// B fragment from a K-contiguous (transposed) source: B[k][n] = Src[n][k].
// Lane l (n=l&15, hi=l>>4): elems 0..7 = K hi*8..hi*8+7, elems 8..15 = +16.
__device__ __forceinline__ v16bf load_bt_frag(const __bf16* p, int ld) {
  const int lane = threadIdx.x & 31;
  const int n = lane & 15;
  const int hi = lane >> 4;
  union { v16bf v; uint4 q[2]; } u;
  const __bf16* col = p + n * ld + hi * 8;
  u.q[0] = *(const uint4*)(col);
  u.q[1] = *(const uint4*)(col + 16);
  return u.v;
}

#if HAS_TDM
// Issue a TDM 2D tile load (bf16 elements) global -> LDS with row padding.
// pad_icode: (1<<pad_icode)*2 dwords between pads; pad_acode: pad_acode+1
// dwords of padding. D# layout per CDNA5 ISA 8.3-8.6.
__device__ __forceinline__ void tdm_load_2d(
    unsigned lds_addr, const void* gaddr,
    unsigned tile_d0, unsigned tile_d1,
    unsigned tensor_d0, unsigned tensor_d1,
    unsigned stride_d0,
    unsigned pad_icode, unsigned pad_acode) {
  const unsigned long long ga = (unsigned long long)(uintptr_t)gaddr;
  v4u g0;
  g0[0] = 1u;                                           // count=1, user D#
  g0[1] = lds_addr;                                     // LDS byte address
  g0[2] = (unsigned)(ga & 0xffffffffu);                 // global_addr[31:0]
  g0[3] = (unsigned)((ga >> 32) & 0x01ffffffu) | (2u << 30);  // addr hi | type=2
  v8i_ g1;
  g1[0] = (int)((1u << 16)            // data_size = 2 bytes
              | (1u << 20)            // pad_enable
              | (pad_icode << 22)     // pad_interval
              | (pad_acode << 25));   // pad_amount
  g1[1] = (int)(tensor_d0 << 16);                              // dim0[15:0]
  g1[2] = (int)((tensor_d0 >> 16) | (tensor_d1 << 16));        // dim0 hi|dim1 lo
  g1[3] = (int)((tensor_d1 >> 16) | (tile_d0 << 16));          // dim1 hi|tile0
  g1[4] = (int)(tile_d1 & 0xffffu);                            // tile1, tile2=0
  g1[5] = (int)stride_d0;                                      // dim0_stride lo
  g1[6] = 0;
  g1[7] = 0;
  const v4i_ gz4 = {0, 0, 0, 0};
  const v8i_ gz8 = {0, 0, 0, 0, 0, 0, 0, 0};
  __builtin_amdgcn_tensor_load_to_lds(g0, g1, gz4, gz4, gz8, 0);
}
#endif

// ---------------------------------------------------------------------------
// Kernel 1: fused QKV projection, register-staged double-buffered pipeline.
// Q/K bf16 [B,H,S,hd]; V bf16 [B,H,hd,S] (transposed for attention's PV).
// ---------------------------------------------------------------------------
__global__ __launch_bounds__(256) void qkv_gemm(
    const float* __restrict__ x,
    const float* __restrict__ wq, const float* __restrict__ wk,
    const float* __restrict__ wv,
    __bf16* __restrict__ qo, __bf16* __restrict__ ko, __bf16* __restrict__ vo) {
  constexpr int TK = 32;
  constexpr int NIT = D / TK;
  __shared__ __align__(16) __bf16 Xs[2][128][TK + 8];   // row-major A tiles
  __shared__ __align__(16) __bf16 Wst[2][128][TK + 8];  // transposed B tiles

  const int tid = threadIdx.x;
  const int n0 = blockIdx.x * 128;
  const int m0 = blockIdx.y * 128;
  const int sel = blockIdx.z;
  const float* w = (sel == 0) ? wq : (sel == 1) ? wk : wv;
  __bf16* outp = (sel == 0) ? qo : (sel == 1) ? ko : vo;
  const float oscale = (sel == 0) ? QSCALE : 1.0f;

  const int wave = tid >> 5;
  const int wm = wave >> 1;   // 0..3 -> 32-row strip
  const int wn = wave & 1;    // 0..1 -> 64-col strip
  const int lane = tid & 31;
  const int ncol = lane & 15;
  const int hi = lane >> 4;

  // Register staging in element pairs along each tile's LDS-contiguous dim:
  // X pairs along k (global b64 loads), W pairs along k (row-strided loads,
  // packed ds_store_b32 into the transposed tile).
  float xr[16], wr[16];
  auto load_regs = [&](int k0) {
#pragma unroll
    for (int j = 0; j < 8; ++j) {
      const int p = tid + j * 256;                 // pair index 0..2047
      const int xrow = p >> 4, xc = (p & 15) * 2;  // 128 rows x 16 pairs
      const float2 xv =
          *(const float2*)(x + (size_t)(m0 + xrow) * D + k0 + xc);
      xr[2 * j] = xv.x;
      xr[2 * j + 1] = xv.y;
      const int wc = p & 127, wrk = (p >> 7) * 2;  // 128 cols x 16 row-pairs
      wr[2 * j]     = w[(size_t)(k0 + wrk)     * D + n0 + wc];
      wr[2 * j + 1] = w[(size_t)(k0 + wrk + 1) * D + n0 + wc];
    }
  };
  auto store_regs = [&](int buf) {
#pragma unroll
    for (int j = 0; j < 8; ++j) {
      const int p = tid + j * 256;
      const int xrow = p >> 4, xc = (p & 15) * 2;
      *(unsigned*)&Xs[buf][xrow][xc] = pk_bf16(xr[2 * j], xr[2 * j + 1]);
      const int wc = p & 127, wrk = (p >> 7) * 2;
      *(unsigned*)&Wst[buf][wc][wrk] = pk_bf16(wr[2 * j], wr[2 * j + 1]);
    }
  };

  const v8f zero = {0.f, 0.f, 0.f, 0.f, 0.f, 0.f, 0.f, 0.f};
  v8f acc[2][4];
#pragma unroll
  for (int i = 0; i < 2; ++i)
#pragma unroll
    for (int j = 0; j < 4; ++j) acc[i][j] = zero;

  load_regs(0);
  store_regs(0);

  for (int it = 0; it < NIT; ++it) {
    const int cur = it & 1;
    const int k0n = (it + 1) * TK;
    if (k0n < D) load_regs(k0n);       // next tile's global loads in flight
    __syncthreads();                   // publish buf[cur]

    v16bf a0 = load_a_frag(&Xs[cur][wm * 32][0], TK + 8);
    v16bf a1 = load_a_frag(&Xs[cur][wm * 32 + 16][0], TK + 8);
#pragma unroll
    for (int nt = 0; nt < 4; ++nt) {
      v16bf b = load_bt_frag(&Wst[cur][wn * 64 + nt * 16][0], TK + 8);
      acc[0][nt] = wmma_bf16(a0, b, acc[0][nt]);
      acc[1][nt] = wmma_bf16(a1, b, acc[1][nt]);
    }
    __syncthreads();                   // everyone done reading buf[cur]
    if (k0n < D) store_regs(1 - cur);  // convert + stage next tile
  }

#pragma unroll
  for (int mt = 0; mt < 2; ++mt) {
#pragma unroll
    for (int nt = 0; nt < 4; ++nt) {
      const int gm = m0 + wm * 32 + mt * 16;
      const int gn = n0 + wn * 64 + nt * 16;
      const int n = gn + ncol;
      const int hh = n >> 7;
      const int dd = n & (HD - 1);
#pragma unroll
      for (int r = 0; r < 8; ++r) {
        const int m = gm + r + 8 * hi;
        const int bb = m >> 11;        // / S
        const int ss = m & (S - 1);
        const __bf16 val = (__bf16)(acc[mt][nt][r] * oscale);
        if (sel == 2)  // V transposed: [B,H,hd,S]
          outp[(((size_t)bb * H + hh) * HD + dd) * S + ss] = val;
        else           // Q,K: [B,H,S,hd]
          outp[(((size_t)bb * H + hh) * S + ss) * HD + dd] = val;
      }
    }
  }
}

// ---------------------------------------------------------------------------
// Kernel 2: causal flash attention. Double-buffered K/V^T tiles streamed by
// the Tensor Data Mover with deferred s_wait_tensorcnt so DMA of tile t+1
// overlaps softmax+WMMA of tile t.
// ---------------------------------------------------------------------------
__global__ __launch_bounds__(256) void attn_kernel(
    const __bf16* __restrict__ Q, const __bf16* __restrict__ K,
    const __bf16* __restrict__ Vt, __bf16* __restrict__ ctx) {
  __shared__ __align__(16) __bf16 Ks[2][KT][HD + 8];    // [key][d]
  __shared__ __align__(16) __bf16 Vst[2][HD][KT + 8];   // [d][key] (transposed)
  __shared__ __align__(16) __bf16 Ps[8][16][KT + 8];

  const int tid = threadIdx.x;
  const int qi = blockIdx.x;
  const int h = blockIdx.y;
  const int b = blockIdx.z;
  const size_t head = (size_t)(b * H + h) * S * HD;
  const __bf16* Qp = Q + head;
  const __bf16* Kp = K + head;
  const __bf16* Vtp = Vt + head;   // per-head [hd][S]

  const int wave = tid >> 5;
  const int lane = tid & 31;
  const int ncol = lane & 15;
  const int hi = lane >> 4;
  const int qrow0 = qi * QT + wave * 16;

  v16bf qa[4];
#pragma unroll
  for (int kk = 0; kk < 4; ++kk)
    qa[kk] = load_a_frag(Qp + (size_t)qrow0 * HD + kk * 32, HD);

  const v8f zero = {0.f, 0.f, 0.f, 0.f, 0.f, 0.f, 0.f, 0.f};
  v8f acc[8];
#pragma unroll
  for (int i = 0; i < 8; ++i) acc[i] = zero;
  float ml[8], ll[8];
#pragma unroll
  for (int r = 0; r < 8; ++r) { ml[r] = -1e30f; ll[r] = 0.f; }

  const int nk = (qi + 1) * (QT / KT);

#if HAS_TDM
  auto issue_tiles = [&](int kt, int buf) {
    const int kb = kt * KT;
    // K tile: 64 rows x 128 bf16, rows of 64 dwords + 4 dword pad
    tdm_load_2d((unsigned)(uintptr_t)&Ks[buf][0][0],
                Kp + (size_t)kb * HD, HD, KT, HD, S, HD, 5, 3);
    // V^T tile: 128 rows x 64 bf16, rows of 32 dwords + 4 dword pad
    tdm_load_2d((unsigned)(uintptr_t)&Vst[buf][0][0],
                Vtp + kb, KT, HD, S, HD, S, 4, 3);
  };
  if (wave == 0) issue_tiles(0, 0);
#endif

  for (int kt = 0; kt < nk; ++kt) {
    const int kb = kt * KT;
    const int cur = kt & 1;
    __syncthreads();  // all waves done with buf[1-cur] (iter kt-1) and Ps
#if HAS_TDM
    if (wave == 0) {
      if (kt + 1 < nk) {
        issue_tiles(kt + 1, 1 - cur);          // DMA next tile concurrently
        __builtin_amdgcn_s_wait_tensorcnt(2);  // in-order: current tile done
      } else {
        __builtin_amdgcn_s_wait_tensorcnt(0);
      }
    }
#else
    for (int t = tid; t < KT * HD; t += 256) {
      const int r = t >> 7, c = t & (HD - 1);
      Ks[cur][r][c] = Kp[(size_t)(kb + r) * HD + c];
    }
    for (int t = tid; t < HD * KT; t += 256) {
      const int r = t >> 6, c = t & (KT - 1);   // r = d, c = key
      Vst[cur][r][c] = Vtp[(size_t)r * S + kb + c];
    }
#endif
    __syncthreads();  // publish buf[cur]

    // S = Q K^T : 4 score tiles of 16x16, k-depth 128 in 4 WMMA steps each.
    v8f sc[4];
#pragma unroll
    for (int nt = 0; nt < 4; ++nt) {
      sc[nt] = zero;
#pragma unroll
      for (int kk = 0; kk < 4; ++kk) {
        v16bf bk = load_bt_frag(&Ks[cur][nt * 16][kk * 32], HD + 8);
        sc[nt] = wmma_bf16(qa[kk], bk, sc[nt]);
      }
    }
    // causal mask
#pragma unroll
    for (int nt = 0; nt < 4; ++nt) {
      const int kg = kb + nt * 16 + ncol;
#pragma unroll
      for (int r = 0; r < 8; ++r) {
        const int qg = qrow0 + r + 8 * hi;
        sc[nt][r] = (kg > qg) ? -1e30f : sc[nt][r];
      }
    }
    // online softmax (row stats per-lane for this half-wave's 8 rows)
    float mnew[8], alpha[8];
#pragma unroll
    for (int r = 0; r < 8; ++r) {
      float v = fmaxf(fmaxf(sc[0][r], sc[1][r]), fmaxf(sc[2][r], sc[3][r]));
      v = fmaxf(v, __shfl_xor(v, 1, 32));
      v = fmaxf(v, __shfl_xor(v, 2, 32));
      v = fmaxf(v, __shfl_xor(v, 4, 32));
      v = fmaxf(v, __shfl_xor(v, 8, 32));
      mnew[r] = fmaxf(ml[r], v);
      alpha[r] = exp2f(ml[r] - mnew[r]);
      ml[r] = mnew[r];
    }
    float rs[8];
#pragma unroll
    for (int r = 0; r < 8; ++r) rs[r] = 0.f;
#pragma unroll
    for (int nt = 0; nt < 4; ++nt)
#pragma unroll
      for (int r = 0; r < 8; ++r) {
        const float p = exp2f(sc[nt][r] - mnew[r]);
        sc[nt][r] = p;
        rs[r] += p;
      }
#pragma unroll
    for (int r = 0; r < 8; ++r) {
      float t = rs[r];
      t += __shfl_xor(t, 1, 32);
      t += __shfl_xor(t, 2, 32);
      t += __shfl_xor(t, 4, 32);
      t += __shfl_xor(t, 8, 32);
      ll[r] = ll[r] * alpha[r] + t;
    }
#pragma unroll
    for (int d = 0; d < 8; ++d)
#pragma unroll
      for (int r = 0; r < 8; ++r) acc[d][r] *= alpha[r];

    // Re-layout P (C-layout regs) -> A fragments via LDS, as bf16.
#pragma unroll
    for (int nt = 0; nt < 4; ++nt)
#pragma unroll
      for (int r = 0; r < 8; ++r)
        Ps[wave][r + 8 * hi][nt * 16 + ncol] = (__bf16)sc[nt][r];
    __syncthreads();

    // O += P @ V  (16x64 @ 64x128); V^T in LDS -> contiguous B fragments.
    v16bf pa0 = load_a_frag(&Ps[wave][0][0], KT + 8);
    v16bf pa1 = load_a_frag(&Ps[wave][0][32], KT + 8);
#pragma unroll
    for (int d = 0; d < 8; ++d) {
      v16bf bv0 = load_bt_frag(&Vst[cur][d * 16][0], KT + 8);
      acc[d] = wmma_bf16(pa0, bv0, acc[d]);
      v16bf bv1 = load_bt_frag(&Vst[cur][d * 16][32], KT + 8);
      acc[d] = wmma_bf16(pa1, bv1, acc[d]);
    }
  }

  // normalize and store ctx[b, s, h*HD + d] as bf16
  __bf16* cp = ctx + (size_t)b * S * D + h * HD;
#pragma unroll
  for (int d = 0; d < 8; ++d)
#pragma unroll
    for (int r = 0; r < 8; ++r) {
      const int m = qrow0 + r + 8 * hi;
      const int dd = d * 16 + ncol;
      cp[(size_t)m * D + dd] = (__bf16)(acc[d][r] / ll[r]);
    }
}

// ---------------------------------------------------------------------------
// Kernel 3: out = ctx @ wo + bo (f32 out). ctx tiles double-buffered via TDM
// with deferred tensorcnt waits; wo register-staged double buffer.
// ---------------------------------------------------------------------------
__global__ __launch_bounds__(256) void out_gemm(
    const __bf16* __restrict__ cx, const float* __restrict__ wo,
    const float* __restrict__ bo, float* __restrict__ out) {
  constexpr int TK = 32;
  constexpr int NIT = D / TK;
  __shared__ __align__(16) __bf16 Xs[2][128][TK + 8];
  __shared__ __align__(16) __bf16 Wst[2][128][TK + 8];

  const int tid = threadIdx.x;
  const int n0 = blockIdx.x * 128;
  const int m0 = blockIdx.y * 128;
  const int wave = tid >> 5;
  const int wm = wave >> 1;
  const int wn = wave & 1;
  const int lane = tid & 31;
  const int ncol = lane & 15;
  const int hi = lane >> 4;

  float wr[16];
  auto load_w_regs = [&](int k0) {
#pragma unroll
    for (int j = 0; j < 8; ++j) {
      const int p = tid + j * 256;
      const int wc = p & 127, wrk = (p >> 7) * 2;
      wr[2 * j]     = wo[(size_t)(k0 + wrk)     * D + n0 + wc];
      wr[2 * j + 1] = wo[(size_t)(k0 + wrk + 1) * D + n0 + wc];
    }
  };
  auto store_w_regs = [&](int buf) {
#pragma unroll
    for (int j = 0; j < 8; ++j) {
      const int p = tid + j * 256;
      const int wc = p & 127, wrk = (p >> 7) * 2;
      *(unsigned*)&Wst[buf][wc][wrk] = pk_bf16(wr[2 * j], wr[2 * j + 1]);
    }
  };
#if !HAS_TDM
  auto stage_x = [&](int k0, int buf) {
    for (int t = tid; t < 128 * TK; t += 256) {
      const int r = t >> 5, c = t & (TK - 1);
      Xs[buf][r][c] = cx[(size_t)(m0 + r) * D + (k0 + c)];
    }
  };
#endif

  const v8f zero = {0.f, 0.f, 0.f, 0.f, 0.f, 0.f, 0.f, 0.f};
  v8f acc[2][4];
#pragma unroll
  for (int i = 0; i < 2; ++i)
#pragma unroll
    for (int j = 0; j < 4; ++j) acc[i][j] = zero;

  load_w_regs(0);
#if HAS_TDM
  if (wave == 0)  // 128 rows x 32 bf16: rows of 16 dwords + 4 dword pad
    tdm_load_2d((unsigned)(uintptr_t)&Xs[0][0][0],
                cx + (size_t)m0 * D, TK, 128, D, M, D, 3, 3);
#else
  stage_x(0, 0);
#endif
  store_w_regs(0);

  for (int it = 0; it < NIT; ++it) {
    const int cur = it & 1;
    const int k0n = (it + 1) * TK;
    if (k0n < D) {
      load_w_regs(k0n);
#if HAS_TDM
      if (wave == 0)
        tdm_load_2d((unsigned)(uintptr_t)&Xs[1 - cur][0][0],
                    cx + (size_t)m0 * D + k0n, TK, 128, D, M, D, 3, 3);
#endif
    }
#if HAS_TDM
    if (wave == 0) {
      if (k0n < D) __builtin_amdgcn_s_wait_tensorcnt(1);  // current tile done
      else         __builtin_amdgcn_s_wait_tensorcnt(0);
    }
#endif
    __syncthreads();  // publish Xs[cur] + Wst[cur]

    v16bf a0 = load_a_frag(&Xs[cur][wm * 32][0], TK + 8);
    v16bf a1 = load_a_frag(&Xs[cur][wm * 32 + 16][0], TK + 8);
#pragma unroll
    for (int nt = 0; nt < 4; ++nt) {
      v16bf b = load_bt_frag(&Wst[cur][wn * 64 + nt * 16][0], TK + 8);
      acc[0][nt] = wmma_bf16(a0, b, acc[0][nt]);
      acc[1][nt] = wmma_bf16(a1, b, acc[1][nt]);
    }
    __syncthreads();  // reads of buf[cur] done
    if (k0n < D) {
      store_w_regs(1 - cur);
#if !HAS_TDM
      stage_x(k0n, 1 - cur);
#endif
    }
  }

#pragma unroll
  for (int mt = 0; mt < 2; ++mt) {
#pragma unroll
    for (int nt = 0; nt < 4; ++nt) {
      const int gm = m0 + wm * 32 + mt * 16;
      const int gn = n0 + wn * 64 + nt * 16;
      const int n = gn + ncol;
      const float bias = bo[n];
#pragma unroll
      for (int r = 0; r < 8; ++r) {
        const int m = gm + r + 8 * hi;
        out[(size_t)m * D + n] = acc[mt][nt][r] + bias;
      }
    }
  }
}

extern "C" void kernel_launch(void* const* d_in, const int* in_sizes, int n_in,
                              void* d_out, int out_size, void* d_ws, size_t ws_size,
                              hipStream_t stream) {
  (void)in_sizes; (void)n_in; (void)out_size; (void)ws_size;
  const float* x  = (const float*)d_in[0];
  const float* wq = (const float*)d_in[1];
  const float* wk = (const float*)d_in[2];
  const float* wv = (const float*)d_in[3];
  const float* wo = (const float*)d_in[4];
  const float* bo = (const float*)d_in[5];
  float* out = (float*)d_out;

  const size_t elems = (size_t)Bz * S * D;  // 8M
  __bf16* qb = (__bf16*)d_ws;
  __bf16* kb = qb + elems;
  __bf16* vb = kb + elems;   // stored transposed [B,H,hd,S]
  __bf16* cb = vb + elems;   // ctx bf16; total 64 MB of workspace

  dim3 blk(256);
  dim3 g1(D / 128, M / 128, 3);
  qkv_gemm<<<g1, blk, 0, stream>>>(x, wq, wk, wv, qb, kb, vb);
  dim3 g2(S / QT, H, Bz);
  attn_kernel<<<g2, blk, 0, stream>>>(qb, kb, vb, cb);
  dim3 g3(D / 128, M / 128);
  out_gemm<<<g3, blk, 0, stream>>>(cb, wo, bo, out);
}
